// Expert_graph_45492293599843
// MI455X (gfx1250) — compile-verified, matrix operand order checked by reading, not run on hardware
//
#include <hip/hip_runtime.h>
#include <hip/hip_bf16.h>

// ---------------------------------------------------------------------------
// MI455X (gfx1250) implementation, round 2.
// Change vs round 1: all WMMA B-operands are pre-packed into fragment-major
// layout (per (n-tile, k-chunk): 32 lanes x 16 bf16 contiguous = 1KB), so a
// B fragment load is two contiguous global_load_b128 instead of 16 strided
// b16 loads + v_mov_b16 packing. A operands stay row-major (already 2x b128).
//
// Pack layout index: P[((nt*KC + kc)*32 + lane)*16 + e]
//   holds element (K = kc*32 + kmap(e, lane>>4), N = nt*16 + (lane&15))
//   kmap(e,hi) = (e<8 ? 0 : 16) + hi*8 + (e&7)      (CDNA5 ISA 7.12.2)
// ---------------------------------------------------------------------------

typedef __bf16 bf16;
typedef bf16  v16bf __attribute__((ext_vector_type(16)));
typedef bf16  v8bf  __attribute__((ext_vector_type(8)));
typedef float v8f   __attribute__((ext_vector_type(8)));

union FragBF { v16bf v; v8bf h[2]; bf16 e[16]; };
union FragF  { v8f   v; float e[8]; };

#define BB   128   // batch
#define LL   128   // sentence length
#define AA   8     // aspect length
#define EE   300   // glove dim
#define EP   320   // padded glove dim (mult of 32)
#define HH   512   // lstm hidden
#define G4   2048  // 4*H
#define NHD  6
#define KAT  64
#define FF   384   // NHD*KAT
#define OUTW 1324  // EE + 2*HH

__device__ __forceinline__ float sigf(float x) { return 1.f / (1.f + __expf(-x)); }

// ---------------------------------------------------------------------------
// Fused transpose+cast+pack: W (N x K row-major fp32) -> fragment-packed bf16
// ---------------------------------------------------------------------------
__global__ void pack_b_kernel(const float* __restrict__ W, bf16* __restrict__ P,
                              int N, int K, int KC) {
  long idx = (long)blockIdx.x * blockDim.x + threadIdx.x;   // over NT*KC*512
  int e    = (int)(idx & 15);
  int lane = (int)((idx >> 4) & 31);
  long blk = idx >> 9;                                      // nt*KC + kc
  int kc = (int)(blk % KC);
  int nt = (int)(blk / KC);
  int n = nt * 16 + (lane & 15);
  int k = kc * 32 + ((e < 8) ? 0 : 16) + (lane >> 4) * 8 + (e & 7);
  float v = (n < N && k < K) ? W[(long)n * K + k] : 0.f;
  P[idx] = (bf16)v;
}

// ---------------------------------------------------------------------------
// Embedding gather + pad + cast (A operand, row-major)
// ---------------------------------------------------------------------------
__global__ void embed_gather_kernel(const int* __restrict__ tok, const float* __restrict__ table,
                                    bf16* __restrict__ X) {
  long row = blockIdx.x;
  int e = threadIdx.x;                     // block = 320
  int t = tok[row];
  float v = (e < EE) ? table[(long)t * EE + e] : 0.f;
  X[row * EP + e] = (bf16)v;
}

// ---------------------------------------------------------------------------
// dep_feature gathered directly into packed-B layout (batched).
// Per batch: NT=EP/16=20 (N = embedding col), KC=LL/32=4 (K = token j).
// ---------------------------------------------------------------------------
__global__ void gather_dep_pack_kernel(const int* __restrict__ dep_tags,
                                       const float* __restrict__ dep_table,
                                       bf16* __restrict__ P) {
  long idx = (long)blockIdx.x * blockDim.x + threadIdx.x;   // BB*20*4*512
  int e    = (int)(idx & 15);
  int lane = (int)((idx >> 4) & 31);
  long blk = idx >> 9;
  int kc = (int)(blk % 4);
  long t2 = blk / 4;
  int nt = (int)(t2 % 20);
  int b  = (int)(t2 / 20);
  int n = nt * 16 + (lane & 15);                            // embedding col
  int k = kc * 32 + ((e < 8) ? 0 : 16) + (lane >> 4) * 8 + (e & 7);  // token j
  int tag = dep_tags[b * LL + k];
  float v = (n < EE) ? dep_table[(long)tag * EE + n] : 0.f;
  P[idx] = (bf16)v;
}

// ---------------------------------------------------------------------------
// Batched WMMA GEMM: C[M,N] = A[M,Kp](bf16,row-major) * Bpack + bias
// One wave per 16x16 tile; block = 128 threads = 4 waves (64 cols / block).
// ---------------------------------------------------------------------------
__global__ void wmma_gemm_kernel(const bf16* __restrict__ A, const bf16* __restrict__ Bp,
                                 const float* __restrict__ bias, float* __restrict__ C,
                                 int M, int N, int Kp, int KC,
                                 long strideA, long strideB, long strideC, int relu) {
  int lane = threadIdx.x & 31;
  int lo = lane & 15, hi = lane >> 4;
  int n0 = (blockIdx.x * (blockDim.x >> 5) + (threadIdx.x >> 5)) * 16;
  int m0 = blockIdx.y * 16;
  if (n0 >= N || m0 >= M) return;          // wave-uniform exit (EXEC stays all-1)

  const bf16* Ab = A + (long)blockIdx.z * strideA + (long)(m0 + lo) * Kp;
  const bf16* Bb = Bp + (long)blockIdx.z * strideB
                      + ((long)(n0 >> 4) * KC) * 512 + (long)lane * 16;
  FragF acc;
  float bv = bias ? bias[n0 + lo] : 0.f;
#pragma unroll
  for (int v = 0; v < 8; ++v) acc.e[v] = bv;

  for (int kc = 0; kc < KC; ++kc) {
    int ka = kc * 32 + hi * 8;
    FragBF a, b;
    a.h[0] = *(const v8bf*)(Ab + ka);       // K = ka .. ka+7
    a.h[1] = *(const v8bf*)(Ab + ka + 16);  // K = ka+16 .. ka+23
    const bf16* p = Bb + (long)kc * 512;
    b.h[0] = *(const v8bf*)(p);
    b.h[1] = *(const v8bf*)(p + 8);
    acc.v = __builtin_amdgcn_wmma_f32_16x16x32_bf16(false, a.v, false, b.v,
                                                    (short)0, acc.v, false, false);
  }
  float* Cb = C + (long)blockIdx.z * strideC;
#pragma unroll
  for (int v = 0; v < 8; ++v) {
    float x = acc.e[v];
    if (relu) x = fmaxf(x, 0.f);
    Cb[(long)(m0 + 8 * hi + v) * N + (n0 + lo)] = x;
  }
}

// ---------------------------------------------------------------------------
// Persistent LSTM scan: one block per 16-row batch tile, all T steps.
// h: LDS bf16 double buffer. c: accumulator registers. WhhP: packed B in L2.
// WhhP pack dims: NT=G4/16=128, KC=HH/32=16.
// ---------------------------------------------------------------------------
__global__ void lstm_scan_kernel(const float* __restrict__ xg, const bf16* __restrict__ WhhP,
                                 float* __restrict__ hs, int T, int reverse,
                                 int out_stride, int out_off) {
  __shared__ bf16 h_smem[2][16][HH];
  int tid = threadIdx.x;
  int lane = tid & 31, w = tid >> 5;       // 8 waves
  int lo = lane & 15, hi = lane >> 4;
  int b0 = blockIdx.x * 16;

  for (int i = tid; i < 16 * HH; i += blockDim.x)
    h_smem[0][i / HH][i % HH] = (bf16)0.f;
  __syncthreads();

  const bf16* Bbase = WhhP + (long)lane * 16;

  FragF creg[4];
#pragma unroll
  for (int t = 0; t < 4; ++t)
#pragma unroll
    for (int v = 0; v < 8; ++v) creg[t].e[v] = 0.f;

  for (int t = 0; t < T; ++t) {
    int tt = reverse ? (T - 1 - t) : t;
    int cur = t & 1, nxt = cur ^ 1;
    for (int ht = 0; ht < 4; ++ht) {
      int j0 = w * 64 + ht * 16;
      FragF acc[4];
#pragma unroll
      for (int q = 0; q < 4; ++q)
#pragma unroll
        for (int v = 0; v < 8; ++v)
          acc[q].e[v] = xg[((long)(b0 + 8 * hi + v) * T + tt) * G4 + q * HH + j0 + lo];

      for (int kc = 0; kc < 16; ++kc) {
        int ka = kc * 32 + hi * 8;
        FragBF a;
        a.h[0] = *(const v8bf*)&h_smem[cur][lo][ka];
        a.h[1] = *(const v8bf*)&h_smem[cur][lo][ka + 16];
#pragma unroll
        for (int q = 0; q < 4; ++q) {
          int nt = (q * HH + j0) >> 4;                 // tile col index
          const bf16* p = Bbase + ((long)(nt * 16 + kc)) * 512;   // KC=16
          FragBF b;
          b.h[0] = *(const v8bf*)(p);
          b.h[1] = *(const v8bf*)(p + 8);
          acc[q].v = __builtin_amdgcn_wmma_f32_16x16x32_bf16(false, a.v, false, b.v,
                                                             (short)0, acc[q].v, false, false);
        }
      }
#pragma unroll
      for (int v = 0; v < 8; ++v) {
        float iv = sigf(acc[0].e[v]);
        float fv = sigf(acc[1].e[v]);
        float gv = tanhf(acc[2].e[v]);
        float ov = sigf(acc[3].e[v]);
        float c  = fv * creg[ht].e[v] + iv * gv;
        creg[ht].e[v] = c;
        float h = ov * tanhf(c);
        int m = 8 * hi + v;
        h_smem[nxt][m][j0 + lo] = (bf16)h;
        hs[((long)(b0 + m) * T + tt) * out_stride + out_off + j0 + lo] = h;
      }
    }
    __syncthreads();
  }
}

// ---------------------------------------------------------------------------
// rel_adj (B,L,L) bf16 (A operand): min(hadj + hadj^T + I, 1) * fmask outer
// ---------------------------------------------------------------------------
__global__ void build_adj_kernel(const int* __restrict__ dep_heads, const int* __restrict__ text_len,
                                 bf16* __restrict__ adj) {
  int j = threadIdx.x, i = blockIdx.x, b = blockIdx.y;
  int tl = text_len[b];
  int hi_ = dep_heads[b * LL + i];
  int hj  = dep_heads[b * LL + j];
  float v = ((hi_ == j) || (hj == i) || (i == j)) ? 1.f : 0.f;
  if (!(i < tl && j < tl)) v = 0.f;
  adj[((long)b * LL + i) * LL + j] = (bf16)v;
}

__global__ void cast_bf16_kernel(const float* __restrict__ src, bf16* __restrict__ dst, long n) {
  long i = (long)blockIdx.x * blockDim.x + threadIdx.x;
  if (i < n) dst[i] = (bf16)src[i];
}

// ---------------------------------------------------------------------------
// score[b,h,l] = q[b,l,h,:] . W2[h,:] + b2[h]; masked softmax over l
// ---------------------------------------------------------------------------
__global__ void rel_score_softmax_kernel(const float* __restrict__ q, const float* __restrict__ W2,
                                         const float* __restrict__ b2, const int* __restrict__ text_len,
                                         float* __restrict__ attn) {
  __shared__ float red[LL];
  int l = threadIdx.x, h = blockIdx.x, b = blockIdx.y;
  const float* qr = q + ((long)(b * LL + l)) * FF + h * KAT;
  float s = b2[h];
  for (int k = 0; k < KAT; ++k) s += qr[k] * W2[h * KAT + k];
  float mask = (l < text_len[b]) ? 1.f : 0.f;
  s = s * mask + (1.f - mask) * (-1e30f);
  red[l] = s; __syncthreads();
  for (int off = 64; off; off >>= 1) { if (l < off) red[l] = fmaxf(red[l], red[l + off]); __syncthreads(); }
  float mx = red[0]; __syncthreads();
  float ex = __expf(s - mx);
  red[l] = ex; __syncthreads();
  for (int off = 64; off; off >>= 1) { if (l < off) red[l] += red[l + off]; __syncthreads(); }
  attn[((long)b * NHD + h) * LL + l] = ex / red[0];
}

// dep_out[b,e] = mean_h relu( sum_l attn[b,h,l] * Hrel[b,l,e] )
__global__ void dep_out_kernel(const float* __restrict__ attn, const float* __restrict__ Hrel,
                               float* __restrict__ out) {
  int e = threadIdx.x, b = blockIdx.x;     // block 320
  if (e >= EE) return;
  float acc = 0.f;
  for (int h = 0; h < NHD; ++h) {
    float s = 0.f;
    const float* aw = attn + ((long)b * NHD + h) * LL;
    for (int l = 0; l < LL; ++l)
      s += aw[l] * Hrel[((long)b * LL + l) * EP + e];
    acc += fmaxf(s, 0.f);
  }
  out[(long)b * OUTW + e] = acc * (1.f / NHD);
}

// aspect mean over T=8
__global__ void asp_mean_kernel(const float* __restrict__ hs_a, float* __restrict__ asp) {
  long i = (long)blockIdx.x * blockDim.x + threadIdx.x;  // B*1024
  if (i >= (long)BB * 2 * HH) return;
  long b = i / (2 * HH); int d = (int)(i % (2 * HH));
  float s = 0.f;
  for (int t = 0; t < AA; ++t) s += hs_a[((long)b * AA + t) * (2 * HH) + d];
  asp[i] = s * (1.f / AA);
}

// dot attention: softmax(feature . asp) over l, gat_out = relu(attn @ feature)
__global__ void dot_attn_kernel(const float* __restrict__ feature, const float* __restrict__ asp,
                                const int* __restrict__ text_len, float* __restrict__ out) {
  __shared__ float av[2 * HH];
  __shared__ float at[LL];
  __shared__ float red[LL];
  int b = blockIdx.x, tid = threadIdx.x;   // block 256
  for (int d = tid; d < 2 * HH; d += 256) av[d] = asp[(long)b * 2 * HH + d];
  __syncthreads();
  float sc = -1e30f;
  if (tid < LL) {
    const float* fr = feature + ((long)b * LL + tid) * (2 * HH);
    float s = 0.f;
    for (int d = 0; d < 2 * HH; ++d) s += fr[d] * av[d];
    float mask = (tid < text_len[b]) ? 1.f : 0.f;
    sc = s * mask + (1.f - mask) * (-1e30f);
  }
  if (tid < LL) red[tid] = sc;
  __syncthreads();
  for (int off = 64; off; off >>= 1) { if (tid < off) red[tid] = fmaxf(red[tid], red[tid + off]); __syncthreads(); }
  float mx = red[0]; __syncthreads();
  float ex = (tid < LL) ? __expf(sc - mx) : 0.f;
  if (tid < LL) red[tid] = ex;
  __syncthreads();
  for (int off = 64; off; off >>= 1) { if (tid < off) red[tid] += red[tid + off]; __syncthreads(); }
  float sum = red[0]; __syncthreads();
  if (tid < LL) at[tid] = ex / sum;
  __syncthreads();
  for (int d = tid; d < 2 * HH; d += 256) {
    float s = 0.f;
    for (int l = 0; l < LL; ++l) s += at[l] * feature[((long)b * LL + l) * (2 * HH) + d];
    out[(long)b * OUTW + EE + d] = fmaxf(s, 0.f);
  }
}

// ---------------------------------------------------------------------------
extern "C" void kernel_launch(void* const* d_in, const int* in_sizes, int n_in,
                              void* d_out, int out_size, void* d_ws, size_t ws_size,
                              hipStream_t stream) {
  const int*   sentence  = (const int*)d_in[0];
  const int*   aspect    = (const int*)d_in[1];
  const int*   dep_tags  = (const int*)d_in[3];
  const int*   text_len  = (const int*)d_in[4];
  const int*   dep_heads = (const int*)d_in[7];
  const float* glove     = (const float*)d_in[10];
  const float* dep_table = (const float*)d_in[11];
  const float* W_ih_f    = (const float*)d_in[12];
  const float* W_hh_f    = (const float*)d_in[13];
  const float* b_f       = (const float*)d_in[14];
  const float* W_ih_b    = (const float*)d_in[15];
  const float* W_hh_b    = (const float*)d_in[16];
  const float* b_b       = (const float*)d_in[17];
  const float* att_W1    = (const float*)d_in[18];
  const float* att_b1    = (const float*)d_in[19];
  const float* att_W2    = (const float*)d_in[20];
  const float* att_b2    = (const float*)d_in[21];
  float* out = (float*)d_out;

  // packed-B sizes (elements): NT*KC*512
  const long szWih = 128L * 10 * 512;   // N=2048 (NT=128), Kp=320 (KC=10)
  const long szWhh = 128L * 16 * 512;   // N=2048, Kp=512 (KC=16)
  const long szW1  = 24L * 10 * 512;    // N=384 (NT=24), Kp=320
  const long szDfe = (long)BB * 20 * 4 * 512;  // per-b: N=320 (NT=20), Kp=128 (KC=4)

  // workspace carve-up
  char* base = (char*)d_ws;
  size_t o = 0;
  auto alloc = [&](size_t bytes) { size_t r = o; o += (bytes + 255) & ~(size_t)255; return r; };
  bf16*  Xs     = (bf16*)(base + alloc((size_t)BB * LL * EP * 2));
  bf16*  Xa     = (bf16*)(base + alloc((size_t)BB * AA * EP * 2));
  bf16*  WihPf  = (bf16*)(base + alloc((size_t)szWih * 2));
  bf16*  WihPb  = (bf16*)(base + alloc((size_t)szWih * 2));
  bf16*  WhhPf  = (bf16*)(base + alloc((size_t)szWhh * 2));
  bf16*  WhhPb  = (bf16*)(base + alloc((size_t)szWhh * 2));
  bf16*  W1P    = (bf16*)(base + alloc((size_t)szW1 * 2));
  float* xg_s   = (float*)(base + alloc((size_t)BB * LL * G4 * 4));   // reused fwd/bwd
  float* xg_a   = (float*)(base + alloc((size_t)BB * AA * G4 * 4));   // reused fwd/bwd
  float* feature= (float*)(base + alloc((size_t)BB * LL * 2 * HH * 4));
  float* hs_a   = (float*)(base + alloc((size_t)BB * AA * 2 * HH * 4));
  float* aspf   = (float*)(base + alloc((size_t)BB * 2 * HH * 4));
  bf16*  adj    = (bf16*)(base + alloc((size_t)BB * LL * LL * 2));
  bf16*  dfeP   = (bf16*)(base + alloc((size_t)szDfe * 2));
  float* Hrel   = (float*)(base + alloc((size_t)BB * LL * EP * 4));
  bf16*  Hrelb  = (bf16*)(base + alloc((size_t)BB * LL * EP * 2));
  float* qbuf   = (float*)(base + alloc((size_t)BB * LL * FF * 4));
  float* attn   = (float*)(base + alloc((size_t)BB * NHD * LL * 4));
  (void)ws_size; (void)n_in; (void)in_sizes; (void)out_size;

  // 1. weight prep (transpose + cast + fragment-pack)
  pack_b_kernel<<<(szWih + 255) / 256, 256, 0, stream>>>(W_ih_f, WihPf, G4, EE, 10);
  pack_b_kernel<<<(szWih + 255) / 256, 256, 0, stream>>>(W_ih_b, WihPb, G4, EE, 10);
  pack_b_kernel<<<(szWhh + 255) / 256, 256, 0, stream>>>(W_hh_f, WhhPf, G4, HH, 16);
  pack_b_kernel<<<(szWhh + 255) / 256, 256, 0, stream>>>(W_hh_b, WhhPb, G4, HH, 16);
  pack_b_kernel<<<(szW1 + 255) / 256, 256, 0, stream>>>(att_W1, W1P, FF, EE, 10);

  // 2. embeddings
  embed_gather_kernel<<<BB * LL, EP, 0, stream>>>(sentence, glove, Xs);
  embed_gather_kernel<<<BB * AA, EP, 0, stream>>>(aspect, glove, Xa);

  // 3-4. sentence BiLSTM (forward then backward, xg buffer reused)
  {
    dim3 g(G4 / 64, BB * LL / 16, 1);
    wmma_gemm_kernel<<<g, 128, 0, stream>>>(Xs, WihPf, b_f, xg_s, BB * LL, G4, EP, 10, 0, 0, 0, 0);
    lstm_scan_kernel<<<BB / 16, 256, 0, stream>>>(xg_s, WhhPf, feature, LL, 0, 2 * HH, 0);
    wmma_gemm_kernel<<<g, 128, 0, stream>>>(Xs, WihPb, b_b, xg_s, BB * LL, G4, EP, 10, 0, 0, 0, 0);
    lstm_scan_kernel<<<BB / 16, 256, 0, stream>>>(xg_s, WhhPb, feature, LL, 1, 2 * HH, HH);
  }
  // aspect BiLSTM (same weights, T=8)
  {
    dim3 g(G4 / 64, BB * AA / 16, 1);
    wmma_gemm_kernel<<<g, 128, 0, stream>>>(Xa, WihPf, b_f, xg_a, BB * AA, G4, EP, 10, 0, 0, 0, 0);
    lstm_scan_kernel<<<BB / 16, 256, 0, stream>>>(xg_a, WhhPf, hs_a, AA, 0, 2 * HH, 0);
    wmma_gemm_kernel<<<g, 128, 0, stream>>>(Xa, WihPb, b_b, xg_a, BB * AA, G4, EP, 10, 0, 0, 0, 0);
    lstm_scan_kernel<<<BB / 16, 256, 0, stream>>>(xg_a, WhhPb, hs_a, AA, 1, 2 * HH, HH);
  }
  asp_mean_kernel<<<(BB * 2 * HH + 255) / 256, 256, 0, stream>>>(hs_a, aspf);

  // 5. Hrel = rel_adj @ dep_feature (batched WMMA; B gathered pre-packed)
  build_adj_kernel<<<dim3(LL, BB), LL, 0, stream>>>(dep_heads, text_len, adj);
  gather_dep_pack_kernel<<<(szDfe + 255) / 256, 256, 0, stream>>>(dep_tags, dep_table, dfeP);
  {
    dim3 g(EP / 64, LL / 16, BB);
    wmma_gemm_kernel<<<g, 128, 0, stream>>>(adj, dfeP, nullptr, Hrel, LL, EP, LL, 4,
                                            (long)LL * LL, 20L * 4 * 512, (long)LL * EP, 0);
  }
  cast_bf16_kernel<<<(BB * LL * EP + 255) / 256, 256, 0, stream>>>(Hrel, Hrelb, (long)BB * LL * EP);

  // 6. q = relu(Hrel @ W1^T + b1)   (M=16384, N=384, K=320)
  {
    dim3 g(FF / 64, BB * LL / 16, 1);
    wmma_gemm_kernel<<<g, 128, 0, stream>>>(Hrelb, W1P, att_b1, qbuf, BB * LL, FF, EP, 10, 0, 0, 0, 1);
  }

  // 7. heads: score + masked softmax, dep_out, dot attention
  rel_score_softmax_kernel<<<dim3(NHD, BB), LL, 0, stream>>>(qbuf, att_W2, att_b2, text_len, attn);
  dep_out_kernel<<<BB, EP, 0, stream>>>(attn, Hrel, out);
  dot_attn_kernel<<<BB, 256, 0, stream>>>(feature, aspf, text_len, out);
}
